// MSC_75453985457573
// MI455X (gfx1250) — matrix operand order repaired and checked by reading, not compiled, not run on hardware
//
#include <hip/hip_runtime.h>
#include <hip/hip_bf16.h>
#include <math.h>

// ---------------------------------------------------------------------------
// Problem constants (from reference)
// ---------------------------------------------------------------------------
#define BATCH 2
#define CCH   256
#define HH    56
#define WW    56
#define NPIX  (HH*WW)          // 3136
#define HEADS 8
#define DHEAD 32
#define BN    (BATCH*NPIX)     // 6272
#define K1SEL 1568             // 3136 // 2
#define K2SEL 1045             // 3136 // 3
#define LNEPS 1e-5f

typedef _Float16 v16h __attribute__((ext_vector_type(16)));
typedef float    v8f  __attribute__((ext_vector_type(8)));
typedef unsigned int v4u __attribute__((ext_vector_type(4)));
typedef float    v4f  __attribute__((ext_vector_type(4)));

union FragH { v16h h; v4u u[2]; };

// ---------------------------------------------------------------------------
// WMMA fragment loaders (CDNA5 16-bit layouts, wave32)
//   A (16x32, MxK), row-major A[M][K]:
//     lane r=lane&15 -> M=r ; K = (lane>=16?8:0) + (i&7) + (i>=8?16:0)
//     -> two contiguous 16B runs per lane.
//   B (32x16, KxN) stored as Bt[N][K] row-major:
//     lane -> N=lane&15 ; K = (lane>=16?8:0) + (i&7) + (i>=8?16:0)  -> two 16B runs.
// ---------------------------------------------------------------------------
__device__ __forceinline__ v16h load_frag_a(const _Float16* A, int lda,
                                            int row0, int k0, int lane) {
  int r  = row0 + (lane & 15);
  int kb = k0 + ((lane >> 4) ? 8 : 0);
  const _Float16* p = A + (size_t)r * lda + kb;
  FragH f;
  f.u[0] = *reinterpret_cast<const v4u*>(p);
  f.u[1] = *reinterpret_cast<const v4u*>(p + 16);
  return f.h;
}

__device__ __forceinline__ v16h load_frag_b(const _Float16* Bt, int ldb,
                                            int col0, int k0, int lane) {
  int c  = col0 + (lane & 15);
  int kb = k0 + ((lane >> 4) ? 8 : 0);
  const _Float16* p = Bt + (size_t)c * ldb + kb;
  FragH f;
  f.u[0] = *reinterpret_cast<const v4u*>(p);
  f.u[1] = *reinterpret_cast<const v4u*>(p + 16);
  return f.h;
}

// ---------------------------------------------------------------------------
// Generic WMMA GEMM, 16x64 output strip per wave (A fragment reused 4x):
//   D = A[M][K] * Bt[NC][K]^T (+bias)
// MODE: 0=Q-proj  1=KV-proj  2=S(QK^T, scaled)  4=final proj (+transpose out)
// ---------------------------------------------------------------------------
template<int MODE>
__global__ __launch_bounds__(256)
void gemm_wmma(const _Float16* __restrict__ A, const _Float16* __restrict__ Bt,
               const float* __restrict__ bias, int M, int NC, int K,
               int lda, int ldb, void* __restrict__ out, void* __restrict__ out2,
               float scale) {
  const int lane = threadIdx.x & 31;
  const int wave = threadIdx.x >> 5;
  const int nt64 = NC >> 6;                       // 64-col strips
  const int tiles = (M >> 4) * nt64;
  int tile = blockIdx.x * (blockDim.x >> 5) + wave;
  if (tile >= tiles) return;
  const int tm  = tile / nt64;
  const int tc0 = (tile - tm * nt64) * 64;        // first output column

  v8f acc[4] = {{}, {}, {}, {}};
  for (int k0 = 0; k0 < K; k0 += 32) {
    v16h a = load_frag_a(A, lda, tm * 16, k0, lane);
    if (k0 + 32 < K) {
      __builtin_prefetch(A + (size_t)(tm * 16 + (lane & 15)) * lda + k0 + 32, 0, 1);
      __builtin_prefetch(Bt + (size_t)(tc0 + (lane & 15)) * ldb + k0 + 32, 0, 1);
    }
#pragma unroll
    for (int t = 0; t < 4; ++t) {
      v16h b = load_frag_b(Bt, ldb, tc0 + t * 16, k0, lane);
      acc[t] = __builtin_amdgcn_wmma_f32_16x16x32_f16(
          false, a, false, b, (short)0, acc[t], false, false);
    }
  }

  const int rbase = tm * 16 + ((lane >> 4) ? 8 : 0);
#pragma unroll
  for (int t = 0; t < 4; ++t) {
    const int col = tc0 + t * 16 + (lane & 15);
    const float bv = bias ? bias[col] : 0.0f;
#pragma unroll
    for (int j = 0; j < 8; ++j) {
      int row = rbase + j;
      float v = acc[t][j] * scale + bv;
      if (MODE == 2) {
        // attention scores: S[row][col], NC == N
        reinterpret_cast<float*>(out)[(size_t)row * NC + col] = v;
      } else if (MODE == 0) {
        // q_h[(b*8+h)*N + n][dd]
        int b = row / NPIX, n = row - b * NPIX;
        int hh = col >> 5, dd = col & 31;
        reinterpret_cast<_Float16*>(out)[
            (((size_t)(b * HEADS + hh)) * NPIX + n) * DHEAD + dd] = (_Float16)v;
      } else if (MODE == 1) {
        int b = row / NPIX, n = row - b * NPIX;
        if (col < CCH) {        // K half -> k_h[(b*8+h)*N + m][dd]
          int hh = col >> 5, dd = col & 31;
          reinterpret_cast<_Float16*>(out)[
              (((size_t)(b * HEADS + hh)) * NPIX + n) * DHEAD + dd] = (_Float16)v;
        } else {                // V half -> vT_h[(b*8+h)*32 + dd][m]
          int o = col - CCH;
          int hh = o >> 5, dd = o & 31;
          reinterpret_cast<_Float16*>(out2)[
              (((size_t)(b * HEADS + hh)) * DHEAD + dd) * (size_t)NPIX + n] = (_Float16)v;
        }
      } else { // MODE == 4, final projection, write transposed to [B][C][H][W]
        int b = row / NPIX, n = row - b * NPIX;
        reinterpret_cast<float*>(out)[
            ((size_t)b * CCH + col) * (size_t)NPIX + n] = v;
      }
    }
  }
}

// ---------------------------------------------------------------------------
// AV GEMM with fused top-k softmax weighting. One wave owns a full 16x32
// output strip (both dd tiles), so the exp-weighted A fragment is built once
// and reused for 2 WMMAs per k-step.
//   P[n,m] = exp(S-mx) * (p1*(S>=thr1)/s1 + p2*(S>=thr2)/s2)
//   out[n,dd] = sum_m P[n,m] * V[m,dd]     (Bt = vT_h slice, [32][N])
// ---------------------------------------------------------------------------
__global__ __launch_bounds__(256)
void av_gemm(const float* __restrict__ S, const float* __restrict__ stats,
             const _Float16* __restrict__ vT, _Float16* __restrict__ attn_out,
             const float* __restrict__ p1p, const float* __restrict__ p2p,
             int bh) {
  const int lane = threadIdx.x & 31;
  const int wave = threadIdx.x >> 5;
  const int tiles = NPIX >> 4;              // 196 row tiles
  int tm = blockIdx.x * (blockDim.x >> 5) + wave;
  if (tm >= tiles) return;

  const int r = tm * 16 + (lane & 15);      // query row (A-fragment row)
  const float thr1 = stats[r * 5 + 0];
  const float thr2 = stats[r * 5 + 1];
  const float mx   = stats[r * 5 + 2];
  const float c1   = p1p[0] / stats[r * 5 + 3];
  const float c2   = p2p[0] / stats[r * 5 + 4];

  v8f acc0 = {}, acc1 = {};
  for (int k0 = 0; k0 < NPIX; k0 += 32) {
    const int kb = k0 + ((lane >> 4) ? 8 : 0);
    const float* sp = S + (size_t)r * NPIX + kb;
    if (k0 + 32 < NPIX) __builtin_prefetch(sp + 32, 0, 1);
    v4f x0 = *reinterpret_cast<const v4f*>(sp);
    v4f x1 = *reinterpret_cast<const v4f*>(sp + 4);
    v4f x2 = *reinterpret_cast<const v4f*>(sp + 16);
    v4f x3 = *reinterpret_cast<const v4f*>(sp + 20);
    float xv[16];
#pragma unroll
    for (int i = 0; i < 4; ++i) {
      xv[i] = x0[i]; xv[4 + i] = x1[i]; xv[8 + i] = x2[i]; xv[12 + i] = x3[i];
    }
    v16h a;
#pragma unroll
    for (int i = 0; i < 16; ++i) {
      float x = xv[i];
      float w = __expf(x - mx) *
                ((x >= thr1 ? c1 : 0.0f) + (x >= thr2 ? c2 : 0.0f));
      a[i] = (_Float16)w;
    }
    v16h b0 = load_frag_b(vT, NPIX, 0, k0, lane);
    v16h b1 = load_frag_b(vT, NPIX, 16, k0, lane);
    acc0 = __builtin_amdgcn_wmma_f32_16x16x32_f16(
        false, a, false, b0, (short)0, acc0, false, false);
    acc1 = __builtin_amdgcn_wmma_f32_16x16x32_f16(
        false, a, false, b1, (short)0, acc1, false, false);
  }

  const int b  = bh >> 3;
  const int hbase = (bh & 7) * DHEAD;
  const int cl = lane & 15;
  const int rbase = tm * 16 + ((lane >> 4) ? 8 : 0);
#pragma unroll
  for (int j = 0; j < 8; ++j) {
    int n = rbase + j;
    _Float16* o = attn_out + ((size_t)b * NPIX + n) * CCH + hbase;
    o[cl]      = (_Float16)acc0[j];
    o[16 + cl] = (_Float16)acc1[j];
  }
}

// ---------------------------------------------------------------------------
// Per-row top-k thresholds via 8-bit radix select on monotonic float keys
// (exact k-th largest, 4 histogram passes per threshold) + softmax sums.
// stats[n] = { thr1, thr2, rowmax, sum1, sum2 }
// ---------------------------------------------------------------------------
__device__ __forceinline__ unsigned fkey(float f) {
  unsigned u = __float_as_uint(f);
  return (u & 0x80000000u) ? ~u : (u | 0x80000000u);   // ascending order key
}
__device__ __forceinline__ float fkey_inv(unsigned k) {
  unsigned u = (k & 0x80000000u) ? (k & 0x7fffffffu) : ~k;
  return __uint_as_float(u);
}

__global__ __launch_bounds__(256)
void rowstats_kernel(const float* __restrict__ S, float* __restrict__ stats) {
  __shared__ float    sf[256];
  __shared__ unsigned hist[256];
  __shared__ unsigned s_prefix;
  __shared__ int      s_krem;
  const int n   = blockIdx.x;
  const int tid = threadIdx.x;
  const float* row = S + (size_t)n * NPIX;

  // row max (top-1) for softmax stabilization
  float mx = -3.4e38f;
  for (int m = tid; m < NPIX; m += 256) mx = fmaxf(mx, row[m]);
  sf[tid] = mx; __syncthreads();
  for (int s = 128; s > 0; s >>= 1) {
    if (tid < s) sf[tid] = fmaxf(sf[tid], sf[tid + s]);
    __syncthreads();
  }
  mx = sf[0];

  // exact k-th largest via MSB-first radix select
  float thr[2];
  const int kv[2] = {K1SEL, K2SEL};
  for (int t = 0; t < 2; ++t) {
    __syncthreads();
    if (tid == 0) { s_prefix = 0u; s_krem = kv[t]; }
    __syncthreads();
    for (int sh = 24; sh >= 0; sh -= 8) {
      hist[tid] = 0u;
      __syncthreads();
      const unsigned pmask = (sh == 24) ? 0u : (0xFFFFFFFFu << (sh + 8));
      const unsigned pref  = s_prefix;
      for (int m = tid; m < NPIX; m += 256) {
        unsigned u = fkey(row[m]);
        if ((u & pmask) == pref) atomicAdd(&hist[(u >> sh) & 255], 1u);
      }
      __syncthreads();
      if (tid == 0) {
        int kr = s_krem;
        unsigned c = 0;
        for (int b2 = 255; b2 >= 0; --b2) {
          unsigned nb = hist[b2];
          if (c + nb >= (unsigned)kr) {
            s_prefix = pref | ((unsigned)b2 << sh);
            s_krem   = kr - (int)c;
            break;
          }
          c += nb;
        }
      }
      __syncthreads();
    }
    thr[t] = fkey_inv(s_prefix);
  }

  // masked softmax denominators
  float s1 = 0.0f, s2 = 0.0f;
  for (int m = tid; m < NPIX; m += 256) {
    float v = row[m];
    float e = __expf(v - mx);
    if (v >= thr[0]) s1 += e;
    if (v >= thr[1]) s2 += e;
  }
  __syncthreads();
  sf[tid] = s1; __syncthreads();
  for (int s = 128; s > 0; s >>= 1) {
    if (tid < s) sf[tid] += sf[tid + s];
    __syncthreads();
  }
  s1 = sf[0]; __syncthreads();
  sf[tid] = s2; __syncthreads();
  for (int s = 128; s > 0; s >>= 1) {
    if (tid < s) sf[tid] += sf[tid + s];
    __syncthreads();
  }
  s2 = sf[0];

  if (tid == 0) {
    stats[n * 5 + 0] = thr[0];
    stats[n * 5 + 1] = thr[1];
    stats[n * 5 + 2] = mx;
    stats[n * 5 + 3] = fmaxf(s1, 1e-30f);
    stats[n * 5 + 4] = fmaxf(s2, 1e-30f);
  }
}

// ---------------------------------------------------------------------------
// Multiscale pooling: sum of avg-pools 3/5/7 (count_include_pad), per plane.
// ---------------------------------------------------------------------------
__global__ __launch_bounds__(256)
void pool_kernel(const float* __restrict__ y, float* __restrict__ ypool) {
  __shared__ float tile[22][23];
  const int bc = blockIdx.y;                    // b*C + c
  const int tx = (blockIdx.x & 3) * 16;
  const int ty = (blockIdx.x >> 2) * 16;
  const float* plane = y + (size_t)bc * NPIX;

  for (int idx = threadIdx.y * 16 + threadIdx.x; idx < 22 * 22; idx += 256) {
    int ly = idx / 22, lx = idx - ly * 22;
    int gy = ty + ly - 3, gx = tx + lx - 3;
    float v = 0.0f;
    if (gy >= 0 && gy < HH && gx >= 0 && gx < WW) v = plane[gy * WW + gx];
    tile[ly][lx] = v;
  }
  __syncthreads();

  int ox = tx + threadIdx.x, oy = ty + threadIdx.y;
  if (ox < WW && oy < HH) {
    float s3 = 0.0f, s5 = 0.0f, s7 = 0.0f;
#pragma unroll
    for (int dy = -3; dy <= 3; ++dy)
#pragma unroll
      for (int dx = -3; dx <= 3; ++dx) {
        float v = tile[threadIdx.y + 3 + dy][threadIdx.x + 3 + dx];
        s7 += v;
        if (dy >= -2 && dy <= 2 && dx >= -2 && dx <= 2) s5 += v;
        if (dy >= -1 && dy <= 1 && dx >= -1 && dx <= 1) s3 += v;
      }
    ypool[(size_t)bc * NPIX + oy * WW + ox] =
        s3 * (1.0f / 9.0f) + s5 * (1.0f / 25.0f) + s7 * (1.0f / 49.0f);
  }
}

// ---------------------------------------------------------------------------
// LayerNorm over C per pixel; writes f16 [B*N][C] (row-major, GEMM A layout)
// ---------------------------------------------------------------------------
__global__ __launch_bounds__(256)
void ln_kernel(const float* __restrict__ ypool, const float* __restrict__ lnw,
               const float* __restrict__ lnb, _Float16* __restrict__ yf) {
  __shared__ float sf[256];
  const int p = blockIdx.x;                 // b*N + n
  const int b = p / NPIX, n = p - b * NPIX;
  const int c = threadIdx.x;
  float v = ypool[((size_t)b * CCH + c) * NPIX + n];

  sf[c] = v; __syncthreads();
  for (int s = 128; s > 0; s >>= 1) {
    if (c < s) sf[c] += sf[c + s];
    __syncthreads();
  }
  float mu = sf[0] * (1.0f / CCH); __syncthreads();
  float d = v - mu;
  sf[c] = d * d; __syncthreads();
  for (int s = 128; s > 0; s >>= 1) {
    if (c < s) sf[c] += sf[c + s];
    __syncthreads();
  }
  float rstd = rsqrtf(sf[0] * (1.0f / CCH) + LNEPS);
  yf[(size_t)p * CCH + c] = (_Float16)(d * rstd * lnw[c] + lnb[c]);
}

// ---------------------------------------------------------------------------
// f32 -> f16 conversions
// ---------------------------------------------------------------------------
__global__ void conv_f16_kernel(const float* __restrict__ in,
                                _Float16* __restrict__ out, int n) {
  int i = blockIdx.x * blockDim.x + threadIdx.x;
  if (i < n) out[i] = (_Float16)in[i];
}

// x [B][C][N] -> x_h [B*N][C] (tiled LDS transpose + convert, coalesced both ways)
__global__ __launch_bounds__(256)
void transpose_x_kernel(const float* __restrict__ x, _Float16* __restrict__ xh) {
  __shared__ _Float16 t[32][33];
  const int b  = blockIdx.z;
  const int n0 = blockIdx.x * 32;
  const int c0 = blockIdx.y * 32;
  const int tx = threadIdx.x & 31;
  const int ty = threadIdx.x >> 5;          // 0..7
#pragma unroll
  for (int i = 0; i < 32; i += 8) {
    int c = c0 + ty + i;
    t[ty + i][tx] = (_Float16)x[((size_t)b * CCH + c) * NPIX + n0 + tx];
  }
  __syncthreads();
#pragma unroll
  for (int i = 0; i < 32; i += 8) {
    int n = n0 + ty + i;
    xh[((size_t)b * NPIX + n) * CCH + c0 + tx] = t[tx][ty + i];
  }
}

// ---------------------------------------------------------------------------
// Host launcher
// ---------------------------------------------------------------------------
extern "C" void kernel_launch(void* const* d_in, const int* in_sizes, int n_in,
                              void* d_out, int out_size, void* d_ws, size_t ws_size,
                              hipStream_t stream) {
  const float* x      = (const float*)d_in[0];
  const float* y      = (const float*)d_in[1];
  const float* q_w    = (const float*)d_in[2];
  const float* q_b    = (const float*)d_in[3];
  const float* kv_w   = (const float*)d_in[4];
  const float* kv_b   = (const float*)d_in[5];
  const float* proj_w = (const float*)d_in[6];
  const float* proj_b = (const float*)d_in[7];
  const float* ln_w   = (const float*)d_in[8];
  const float* ln_b   = (const float*)d_in[9];
  const float* p1     = (const float*)d_in[10];
  const float* p2     = (const float*)d_in[11];

  // workspace layout
  char* ws = (char*)d_ws;
  size_t off = 0;
  auto alloc = [&](size_t bytes) -> char* {
    char* p = ws + off;
    off = (off + bytes + 255) & ~(size_t)255;
    return p;
  };
  _Float16* x_h   = (_Float16*)alloc((size_t)BN * CCH * 2);
  _Float16* yf_h  = (_Float16*)alloc((size_t)BN * CCH * 2);
  _Float16* qw_h  = (_Float16*)alloc((size_t)CCH * CCH * 2);
  _Float16* kvw_h = (_Float16*)alloc((size_t)2 * CCH * CCH * 2);
  _Float16* pw_h  = (_Float16*)alloc((size_t)CCH * CCH * 2);
  _Float16* q_h   = (_Float16*)alloc((size_t)BN * CCH * 2);
  _Float16* k_h   = (_Float16*)alloc((size_t)BN * CCH * 2);
  _Float16* vT_h  = (_Float16*)alloc((size_t)BN * CCH * 2);
  _Float16* ao_h  = (_Float16*)alloc((size_t)BN * CCH * 2);
  float*    ypool = (float*)alloc((size_t)BATCH * CCH * NPIX * 4);
  float*    Sbuf  = (float*)alloc((size_t)NPIX * NPIX * 4);
  float*    stats = (float*)alloc((size_t)NPIX * 5 * 4);

  // 1) input conversions
  transpose_x_kernel<<<dim3(NPIX / 32, CCH / 32, BATCH), 256, 0, stream>>>(x, x_h);
  conv_f16_kernel<<<(CCH * CCH + 255) / 256, 256, 0, stream>>>(q_w, qw_h, CCH * CCH);
  conv_f16_kernel<<<(2 * CCH * CCH + 255) / 256, 256, 0, stream>>>(kv_w, kvw_h, 2 * CCH * CCH);
  conv_f16_kernel<<<(CCH * CCH + 255) / 256, 256, 0, stream>>>(proj_w, pw_h, CCH * CCH);

  // 2) multiscale pool + layernorm -> yf_h
  pool_kernel<<<dim3(16, BATCH * CCH), dim3(16, 16), 0, stream>>>(y, ypool);
  ln_kernel<<<BN, 256, 0, stream>>>(ypool, ln_w, ln_b, yf_h);

  // 3) q / kv projections (WMMA)
  {
    int tiles = (BN / 16) * (CCH / 64);
    gemm_wmma<0><<<(tiles + 7) / 8, 256, 0, stream>>>(
        x_h, qw_h, q_b, BN, CCH, CCH, CCH, CCH, q_h, nullptr, 1.0f);
  }
  {
    int tiles = (BN / 16) * (2 * CCH / 64);
    gemm_wmma<1><<<(tiles + 7) / 8, 256, 0, stream>>>(
        yf_h, kvw_h, kv_b, BN, 2 * CCH, CCH, CCH, CCH, k_h, vT_h, 1.0f);
  }

  // 4) attention per (b, h): S = QK^T*scale ; row thresholds ; out = P V
  const float scale = 1.0f / sqrtf((float)DHEAD);
  for (int bh = 0; bh < BATCH * HEADS; ++bh) {
    const _Float16* qs = q_h + (size_t)bh * NPIX * DHEAD;
    const _Float16* ks = k_h + (size_t)bh * NPIX * DHEAD;
    const _Float16* vs = vT_h + (size_t)bh * DHEAD * NPIX;
    int tiles = (NPIX / 16) * (NPIX / 64);
    gemm_wmma<2><<<(tiles + 7) / 8, 256, 0, stream>>>(
        qs, ks, nullptr, NPIX, NPIX, DHEAD, DHEAD, DHEAD, Sbuf, nullptr, scale);
    rowstats_kernel<<<NPIX, 256, 0, stream>>>(Sbuf, stats);
    av_gemm<<<(NPIX / 16 + 7) / 8, 256, 0, stream>>>(
        Sbuf, stats, vs, ao_h, p1, p2, bh);
  }

  // 5) final projection + transpose to [B][C][H][W]
  {
    int tiles = (BN / 16) * (CCH / 64);
    gemm_wmma<4><<<(tiles + 7) / 8, 256, 0, stream>>>(
        ao_h, pw_h, proj_b, BN, CCH, CCH, CCH, CCH, d_out, nullptr, 1.0f);
  }
}
